// BahdanauAttention_8504035246532
// MI455X (gfx1250) — compile-verified, matrix operand order checked by reading, not run on hardware
//
#include <hip/hip_runtime.h>
#include <hip/hip_bf16.h>
#include <math.h>

typedef __attribute__((ext_vector_type(16))) _Float16 v16h;
typedef __attribute__((ext_vector_type(8)))  float    v8f;

#define BB 64
#define NN 1024
#define DD 1024

struct alignas(8) H4 { _Float16 h[4]; };

// Branch-free tanh: hardware v_tanh_f32 on gfx1250 if exposed, else exp-based.
__device__ __forceinline__ float fast_tanh(float x) {
#if __has_builtin(__builtin_amdgcn_tanhf)
    return __builtin_amdgcn_tanhf(x);
#else
    // 1 - 2/(e^{2x}+1): saturates to +/-1 at large |x|, no branches.
    float ex = __expf(2.0f * x);
    return 1.0f - 2.0f / (ex + 1.0f);
#endif
}

// ---------------------------------------------------------------------------
// Kernel 1: qh[b,e] = sum_d h_t[b,d] * W_h[e,d]
// ---------------------------------------------------------------------------
__global__ __launch_bounds__(256) void qh_kernel(const float* __restrict__ h_t,
                                                 const float* __restrict__ W_h,
                                                 float* __restrict__ qh) {
    const int b = blockIdx.x;
    const int t = threadIdx.x;
    __shared__ float hrow[DD];
    for (int d = t; d < DD; d += 256) hrow[d] = h_t[(size_t)b * DD + d];
    __syncthreads();

    float acc[4] = {0.f, 0.f, 0.f, 0.f};
    for (int d4 = 0; d4 < DD / 4; ++d4) {
        float4 h4 = ((const float4*)hrow)[d4];
#pragma unroll
        for (int j = 0; j < 4; ++j) {
            int e = t + j * 256;
            float4 w4 = ((const float4*)(W_h + (size_t)e * DD))[d4];
            acc[j] += h4.x * w4.x + h4.y * w4.y + h4.z * w4.z + h4.w * w4.w;
        }
    }
#pragma unroll
    for (int j = 0; j < 4; ++j) qh[(size_t)b * DD + t + j * 256] = acc[j];
}

// ---------------------------------------------------------------------------
// Kernel 2: convert W_k (fp32 row-major [e][d]) to f16 in B-fragment layout:
//   dst half index = ((et*32 + kc)*32 + lane)*16 + i
//   where e = et*16 + (lane&15), d = kc*32 + (i&7) + 8*(lane>>4) + 16*(i>>3)
// ---------------------------------------------------------------------------
__global__ __launch_bounds__(256) void wk_convert(const float* __restrict__ W_k,
                                                  _Float16* __restrict__ wkB) {
    const int gid = blockIdx.x * 256 + threadIdx.x;   // each thread: 4 halfs
    const int o    = gid * 4;                         // dst half index
    const int i0   = o & 15;                          // 0,4,8,12
    const int lane = (o >> 4) & 31;
    const int kc   = (o >> 9) & 31;
    const int et   = (o >> 14);
    const int e    = et * 16 + (lane & 15);
    const int h    = lane >> 4;
    const int j    = i0 >> 3;
    const int d0   = kc * 32 + (i0 & 7) + 8 * h + 16 * j;
    float4 w = *(const float4*)(W_k + (size_t)e * DD + d0);
    H4 out;
    out.h[0] = (_Float16)w.x; out.h[1] = (_Float16)w.y;
    out.h[2] = (_Float16)w.z; out.h[3] = (_Float16)w.w;
    *(H4*)(wkB + o) = out;
}

// ---------------------------------------------------------------------------
// Kernel 3: fused scores.  WG = 64 rows (one batch), 4 waves x 16-row subtile.
// Dynamic LDS: keys tile as f16 in A-fragment layout, [4][32][32][16] halfs.
// scores[b,n] = sum_e tanh(kh[n,e] + qh[b,e]) * v[e]
// qh is folded into the WMMA C initializer (column e constant per lane).
// ---------------------------------------------------------------------------
__global__ __launch_bounds__(128) void attn_scores(const float* __restrict__ keys,
                                                   const _Float16* __restrict__ wkB,
                                                   const float* __restrict__ qh,
                                                   const float* __restrict__ v,
                                                   float* __restrict__ scores) {
    extern __shared__ _Float16 keysA[];               // 65536 halfs = 128 KB
    const int tid = threadIdx.x;
    const int m0  = blockIdx.x * 64;                  // global row base
    const int b   = m0 >> 10;                         // 64 | 1024 -> same batch
    const int n0  = m0 & (NN - 1);

    // ---- Phase 1: stage keys tile (64x1024 fp32 -> f16 A-fragment layout)
    for (int q = tid; q < 64 * (DD / 4); q += 128) {
        const int m = q >> 8;                         // row in tile
        const int d = (q & 255) * 4;
        float4 kv = *(const float4*)(keys + (size_t)(m0 + m) * DD + d);
        const int mt   = m >> 4;
        const int kc   = d >> 5;
        const int dd   = d & 31;
        const int h    = (dd >> 3) & 1;
        const int lane = (m & 15) + (h << 4);
        const int i    = (dd & 7) + ((dd >> 4) << 3);
        H4 hv;
        hv.h[0] = (_Float16)kv.x; hv.h[1] = (_Float16)kv.y;
        hv.h[2] = (_Float16)kv.z; hv.h[3] = (_Float16)kv.w;
        *(H4*)(keysA + ((((mt * 32 + kc) * 32) + lane) * 16 + i)) = hv;
    }
    __syncthreads();

    // ---- Phase 2: WMMA main loop
    const int lane = tid & 31;
    const int wid  = tid >> 5;                        // wave id == M-subtile
    const _Float16* aBase = keysA + (size_t)wid * 32 * 32 * 16;
    const int ecol = lane & 15;

    float s[8];
#pragma unroll
    for (int r = 0; r < 8; ++r) s[r] = 0.f;

    for (int et = 0; et < 64; ++et) {
        const int e = et * 16 + ecol;
        const float qv = qh[(size_t)b * DD + e];
        const float vv = v[e];
        v8f acc = {qv, qv, qv, qv, qv, qv, qv, qv};   // fold +qh into C
        const _Float16* bBase = wkB + (((size_t)et * 32) * 32 + lane) * 16;
#pragma unroll 4
        for (int kc = 0; kc < 32; ++kc) {
            v16h afrag = *(const v16h*)(aBase + (((kc * 32) + lane) << 4));
            v16h bfrag = *(const v16h*)(bBase + ((size_t)kc << 9));
            acc = __builtin_amdgcn_wmma_f32_16x16x32_f16(
                false, afrag, false, bfrag, (short)0, acc, false, false);
        }
#pragma unroll
        for (int r = 0; r < 8; ++r) s[r] = fmaf(fast_tanh(acc[r]), vv, s[r]);
    }

    // ---- Phase 3: reduce partial scores over the 16 e-lanes of each group
#pragma unroll
    for (int r = 0; r < 8; ++r) {
        float x = s[r];
        x += __shfl_xor(x, 1, 16);
        x += __shfl_xor(x, 2, 16);
        x += __shfl_xor(x, 4, 16);
        x += __shfl_xor(x, 8, 16);
        s[r] = x;
    }
    if (ecol == 0) {
        const int half = lane >> 4;                   // rows r + 8*half
#pragma unroll
        for (int r = 0; r < 8; ++r) {
            const int m = r + 8 * half;
            scores[(size_t)b * NN + n0 + wid * 16 + m] = s[r];
        }
    }
}

// ---------------------------------------------------------------------------
// Kernel 4: softmax over N + context = alpha @ keys.  out = [context | alpha]
// ---------------------------------------------------------------------------
__global__ __launch_bounds__(256) void softmax_context(const float* __restrict__ keys,
                                                       const float* __restrict__ scores,
                                                       float* __restrict__ out) {
    const int b = blockIdx.x;
    const int t = threadIdx.x;
    __shared__ float sc[NN];
    __shared__ float red[8];

    for (int n = t; n < NN; n += 256) sc[n] = scores[(size_t)b * NN + n];
    __syncthreads();

    // max
    float mx = -1e30f;
    for (int n = t; n < NN; n += 256) mx = fmaxf(mx, sc[n]);
#pragma unroll
    for (int off = 16; off >= 1; off >>= 1) mx = fmaxf(mx, __shfl_xor(mx, off, 32));
    if ((t & 31) == 0) red[t >> 5] = mx;
    __syncthreads();
    if (t == 0) {
        float m2 = red[0];
#pragma unroll
        for (int w = 1; w < 8; ++w) m2 = fmaxf(m2, red[w]);
        red[0] = m2;
    }
    __syncthreads();
    mx = red[0];
    __syncthreads();

    // exp + sum
    float sum = 0.f;
    for (int n = t; n < NN; n += 256) {
        float e = __expf(sc[n] - mx);
        sc[n] = e;
        sum += e;
    }
#pragma unroll
    for (int off = 16; off >= 1; off >>= 1) sum += __shfl_xor(sum, off, 32);
    if ((t & 31) == 0) red[t >> 5] = sum;
    __syncthreads();
    if (t == 0) {
        float s2 = 0.f;
#pragma unroll
        for (int w = 0; w < 8; ++w) s2 += red[w];
        red[0] = s2;
    }
    __syncthreads();
    const float inv = 1.f / red[0];
    __syncthreads();

    for (int n = t; n < NN; n += 256) {
        float a = sc[n] * inv;
        sc[n] = a;
        out[(size_t)BB * DD + (size_t)b * NN + n] = a;   // alpha
    }
    __syncthreads();

    // context: thread t owns d = 4t..4t+3 (one b128 per row)
    float acc0 = 0.f, acc1 = 0.f, acc2 = 0.f, acc3 = 0.f;
#pragma unroll 4
    for (int n = 0; n < NN; ++n) {
        const float a = sc[n];
        float4 k4 = *(const float4*)(keys + ((size_t)b * NN + n) * DD + t * 4);
        acc0 += a * k4.x; acc1 += a * k4.y; acc2 += a * k4.z; acc3 += a * k4.w;
    }
    float4 c4 = make_float4(acc0, acc1, acc2, acc3);
    *(float4*)(out + (size_t)b * DD + t * 4) = c4;       // context
}

// ---------------------------------------------------------------------------
extern "C" void kernel_launch(void* const* d_in, const int* in_sizes, int n_in,
                              void* d_out, int out_size, void* d_ws, size_t ws_size,
                              hipStream_t stream) {
    const float* h_t  = (const float*)d_in[0];
    const float* keys = (const float*)d_in[1];
    const float* W_h  = (const float*)d_in[2];
    const float* W_k  = (const float*)d_in[3];
    const float* v    = (const float*)d_in[4];
    float* out = (float*)d_out;

    // workspace layout: qh (256KB) | scores (256KB) | wkB f16 (2MB)
    float*     qh     = (float*)d_ws;
    float*     scores = qh + BB * DD;
    _Float16*  wkB    = (_Float16*)(scores + BB * NN);

    qh_kernel<<<BB, 256, 0, stream>>>(h_t, W_h, qh);
    wk_convert<<<(DD * DD / 4) / 256, 256, 0, stream>>>(W_k, wkB);
    attn_scores<<<(BB * NN) / 64, 128, 64 * DD * sizeof(_Float16), stream>>>(
        keys, wkB, qh, v, scores);
    softmax_context<<<BB, 256, 0, stream>>>(keys, scores, out);
}
